// HPTModel_3874060501346
// MI455X (gfx1250) — matrix operand order repaired and checked by reading, not compiled, not run on hardware
//
#include <hip/hip_runtime.h>

// ---------------------------------------------------------------------------
// Problem constants (from reference): B=2, S=4096, H=12, DH=64, W=256, G=64,
// M=256, D=768, C=S/W=16.  Band keys per query chunk = 3W=768, +G global.
// ---------------------------------------------------------------------------
#define Bb   2
#define Ss   4096
#define Hh   12
#define DHd  64
#define Ww   256
#define Gg   64
#define Mm   256
#define Dd   768
#define Cc   16
#define NEGV (-1e9f)

typedef __attribute__((ext_vector_type(16))) _Float16 v16h;
typedef __attribute__((ext_vector_type(8)))  float    v8f;

__device__ __forceinline__ v8f wmma32(v16h a, v16h b, v8f c) {
  return __builtin_amdgcn_wmma_f32_16x16x32_f16(
      /*neg_a=*/false, a, /*neg_b=*/false, b,
      /*c_mod=*/(short)0, c, /*reuse_a=*/false, /*reuse_b=*/false);
}

// A-fragment (16x32 f16, row-major src, leading dim lda in elements).
// CDNA5 layout: lane row = lane&15; k = 8*half + e (e<8), 16 + 8*half + (e-8).
__device__ __forceinline__ v16h load_afrag(const _Float16* A, int lda) {
  int lane = threadIdx.x & 31;
  int row  = lane & 15, half = lane >> 4;
  const _Float16* p = A + (size_t)row * lda + 8 * half;
  v16h a;
#pragma unroll
  for (int e = 0; e < 8; ++e) a[e] = p[e];
#pragma unroll
  for (int e = 0; e < 8; ++e) a[8 + e] = p[16 + e];
  return a;
}

// B-fragment from N x K ("transposed") storage: lane col fixed, k contiguous.
__device__ __forceinline__ v16h load_bfrag_t(const _Float16* Wt, int K,
                                             int ncol, int kk) {
  int lane = threadIdx.x & 31;
  int col = lane & 15, half = lane >> 4;
  const _Float16* p = Wt + (size_t)(ncol + col) * K + kk + 8 * half;
  v16h b;
#pragma unroll
  for (int e = 0; e < 16; ++e) b[e] = p[(e & 7) + ((e >> 3) << 4)];
  return b;
}

__device__ __forceinline__ float sigmoidf_(float x) {
  return 1.f / (1.f + __expf(-x));
}

// ---------------------------------------------------------------------------
// f16 WMMA GEMM: C[M,N] = act(A[M,K] * Wt^T + bias (+residual)),
// Wt stored N x K (pre-transposed) -> fully contiguous B fragments.
// One wave computes a 32(M) x 64(N) tile: 2 A-frags x 4 B-frags = 8 wmma/kstep.
// Requires M % 32 == 0, N % 64 == 0, K % 32 == 0.
// ---------------------------------------------------------------------------
__global__ __launch_bounds__(256) void gemm_f16_wmma(
    const _Float16* __restrict__ A, const _Float16* __restrict__ Wt,
    const float* __restrict__ bias, const float* __restrict__ residual,
    float* __restrict__ Cf, _Float16* __restrict__ Ch,
    int M, int N, int K, int act) {
  int gwave  = (int)((blockIdx.x * 256u + threadIdx.x) >> 5);
  int tilesN = N >> 6;                       // 64-wide N tiles
  int tm = gwave / tilesN, tn = gwave % tilesN;
  if (tm * 32 >= M) return;
  int lane = threadIdx.x & 31, col = lane & 15, half = lane >> 4;

  v8f acc0[4] = {v8f{}, v8f{}, v8f{}, v8f{}};
  v8f acc1[4] = {v8f{}, v8f{}, v8f{}, v8f{}};
  const _Float16* Ap0 = A + (size_t)(tm * 32)      * K;
  const _Float16* Ap1 = A + (size_t)(tm * 32 + 16) * K;
  for (int kk = 0; kk < K; kk += 32) {
    if (kk + 32 < K) {  // prefetch next k-step (global_prefetch_b8)
      __builtin_prefetch(Ap0 + kk + 32, 0, 1);
      __builtin_prefetch(Wt + (size_t)(tn * 64 + col) * K + kk + 32, 0, 1);
    }
    v16h a0 = load_afrag(Ap0 + kk, K);
    v16h a1 = load_afrag(Ap1 + kk, K);
#pragma unroll
    for (int j = 0; j < 4; ++j) {
      v16h b = load_bfrag_t(Wt, K, tn * 64 + j * 16, kk);
      acc0[j] = wmma32(a0, b, acc0[j]);
      acc1[j] = wmma32(a1, b, acc1[j]);
    }
  }
#pragma unroll
  for (int sub = 0; sub < 2; ++sub) {
#pragma unroll
    for (int j = 0; j < 4; ++j) {
#pragma unroll
      for (int g = 0; g < 8; ++g) {
        int row = tm * 32 + sub * 16 + g + 8 * half;
        int cc  = tn * 64 + j * 16 + col;
        float v = (sub == 0) ? acc0[j][g] : acc1[j][g];
        if (bias)     v += bias[cc];
        if (residual) v += residual[(size_t)row * N + cc];
        if (act == 1) v = fmaxf(v, 0.f);
        else if (act == 2) v = sigmoidf_(v);
        if (Cf) Cf[(size_t)row * N + cc] = v;
        if (Ch) Ch[(size_t)row * N + cc] = (_Float16)v;
      }
    }
  }
}

// ---------------------------------------------------------------------------
// Banded + global-key attention (flash-style online softmax).
// Block = (b, h, chunk). 8 waves x 2 query-tiles of 16. 26 key steps of 32:
// steps 0-1 = G global keys, steps 2-25 = 3W band keys.
// ---------------------------------------------------------------------------
__global__ __launch_bounds__(256) void attn_band_kernel(
    const _Float16* __restrict__ qb, const _Float16* __restrict__ kb,
    const _Float16* __restrict__ vb, const float* __restrict__ mask,
    _Float16* __restrict__ ctx16) {
  int blk = blockIdx.x;
  int c = blk % Cc;
  int h = (blk / Cc) % Hh;
  int b = blk / (Cc * Hh);
  int wave = threadIdx.x >> 5, lane = threadIdx.x & 31;
  int col = lane & 15, half = lane >> 4;

  __shared__ _Float16 pbuf[8][16 * 32];
  _Float16* P = pbuf[wave];

  const _Float16* Kbh = kb + (size_t)b * Ss * Dd + h * DHd;
  const _Float16* Vbh = vb + (size_t)b * Ss * Dd + h * DHd;
  const float*    mb  = mask + (size_t)b * Ss;

  for (int tile = wave; tile < 16; tile += 8) {
    int q0 = c * Ww + tile * 16;
    const _Float16* Qp = qb + ((size_t)b * Ss + q0) * Dd + h * DHd;
    v16h qa0 = load_afrag(Qp, Dd);        // dh 0..31
    v16h qa1 = load_afrag(Qp + 32, Dd);   // dh 32..63

    v8f o0 = {}, o1 = {}, o2 = {}, o3 = {};
    float mrun[8], lrun[8];
#pragma unroll
    for (int g = 0; g < 8; ++g) { mrun[g] = -1e30f; lrun[g] = 0.f; }

    for (int step = 0; step < 26; ++step) {
      int jl0 = step * 32 + col;          // linear key index, frag 0
      int jl1 = jl0 + 16;                 // frag 1
      int kp0 = (jl0 < Gg) ? jl0 : (c * Ww - Ww + (jl0 - Gg));
      int kp1 = (jl1 < Gg) ? jl1 : (c * Ww - Ww + (jl1 - Gg));
      int kc0 = min(max(kp0, 0), Ss - 1);
      int kc1 = min(max(kp1, 0), Ss - 1);

      // scores: S = Q (16xdh) * K^T (dhxkeys)
      v8f s0 = {}, s1 = {};
      {
        v16h bk;
        const _Float16* p = Kbh + (size_t)kc0 * Dd + 8 * half;
#pragma unroll
        for (int e = 0; e < 16; ++e) bk[e] = p[(e & 7) + ((e >> 3) << 4)];
        s0 = wmma32(qa0, bk, s0);
#pragma unroll
        for (int e = 0; e < 16; ++e) bk[e] = p[32 + (e & 7) + ((e >> 3) << 4)];
        s0 = wmma32(qa1, bk, s0);
        const _Float16* p1 = Kbh + (size_t)kc1 * Dd + 8 * half;
#pragma unroll
        for (int e = 0; e < 16; ++e) bk[e] = p1[(e & 7) + ((e >> 3) << 4)];
        s1 = wmma32(qa0, bk, s1);
#pragma unroll
        for (int e = 0; e < 16; ++e) bk[e] = p1[32 + (e & 7) + ((e >> 3) << 4)];
        s1 = wmma32(qa1, bk, s1);
      }

      // mask + online softmax update (rows striped g+8*half, cols = lane&15)
      float mv0 = mb[kc0], mv1 = mb[kc1];
#pragma unroll
      for (int g = 0; g < 8; ++g) {
        int i_loc = tile * 16 + g + 8 * half;   // query index within chunk
        float sv0 = s0[g] * 0.125f, sv1 = s1[g] * 0.125f;
        bool ok0, ok1;
        if (jl0 < Gg) ok0 = (mv0 > 0.f);
        else {
          int j = jl0 - Gg, d = j - Ww - i_loc;
          ok0 = (kp0 >= Gg) && (kp0 < Ss) && (d <= Ww) && (d >= -Ww) && (mv0 > 0.f);
        }
        if (jl1 < Gg) ok1 = (mv1 > 0.f);
        else {
          int j = jl1 - Gg, d = j - Ww - i_loc;
          ok1 = (kp1 >= Gg) && (kp1 < Ss) && (d <= Ww) && (d >= -Ww) && (mv1 > 0.f);
        }
        if (!ok0) sv0 = NEGV;
        if (!ok1) sv1 = NEGV;
        float tmax = fmaxf(sv0, sv1);
#pragma unroll
        for (int off = 8; off > 0; off >>= 1)
          tmax = fmaxf(tmax, __shfl_xor(tmax, off, 32));
        float mnew = fmaxf(mrun[g], tmax);
        float sc = __expf(mrun[g] - mnew);
        float p0 = __expf(sv0 - mnew), p1 = __expf(sv1 - mnew);
        float rs = p0 + p1;
#pragma unroll
        for (int off = 8; off > 0; off >>= 1) rs += __shfl_xor(rs, off, 32);
        lrun[g] = lrun[g] * sc + rs;
        mrun[g] = mnew;
        o0[g] *= sc; o1[g] *= sc; o2[g] *= sc; o3[g] *= sc;
        int prow = g + 8 * half;
        P[prow * 32 + col]      = (_Float16)p0;
        P[prow * 32 + 16 + col] = (_Float16)p1;
      }
      asm volatile("s_wait_dscnt 0" ::: "memory");  // LDS transpose visibility

      // P (16x32) as A-fragment
      v16h pa;
      {
        const _Float16* pp = P + (lane & 15) * 32 + 8 * half;
#pragma unroll
        for (int e = 0; e < 8; ++e) pa[e] = pp[e];
#pragma unroll
        for (int e = 0; e < 8; ++e) pa[8 + e] = pp[16 + e];
      }
      // V (32 keys x 64 dh) as 4 B-fragments; masked keys contribute 0 via P=0
      v16h v0, v1, v2, v3;
#pragma unroll
      for (int e = 0; e < 16; ++e) {
        int jl = step * 32 + 8 * half + (e & 7) + ((e >> 3) << 4);
        int kp = (jl < Gg) ? jl : (c * Ww - Ww + (jl - Gg));
        kp = min(max(kp, 0), Ss - 1);
        const _Float16* vp = Vbh + (size_t)kp * Dd + col;
        v0[e] = vp[0]; v1[e] = vp[16]; v2[e] = vp[32]; v3[e] = vp[48];
      }
      o0 = wmma32(pa, v0, o0);
      o1 = wmma32(pa, v1, o1);
      o2 = wmma32(pa, v2, o2);
      o3 = wmma32(pa, v3, o3);
    }

#pragma unroll
    for (int g = 0; g < 8; ++g) {
      float inv = (lrun[g] > 0.f) ? 1.f / lrun[g] : 0.f;
      int srow = q0 + g + 8 * half;
      _Float16* op = ctx16 + ((size_t)b * Ss + srow) * Dd + h * DHd;
      op[col]      = (_Float16)(o0[g] * inv);
      op[16 + col] = (_Float16)(o1[g] * inv);
      op[32 + col] = (_Float16)(o2[g] * inv);
      op[48 + col] = (_Float16)(o3[g] * inv);
    }
  }
}

// ---------------------------------------------------------------------------
// Global queries (rows 0..G-1) attend to all S keys. Block=(b,h), 4 waves,
// one 16-query tile per wave, 128 key steps. Overwrites ctx rows < G.
// ---------------------------------------------------------------------------
__global__ __launch_bounds__(128) void attn_global_kernel(
    const _Float16* __restrict__ qb, const _Float16* __restrict__ kb,
    const _Float16* __restrict__ vb, const float* __restrict__ mask,
    _Float16* __restrict__ ctx16) {
  int h = blockIdx.x % Hh;
  int b = blockIdx.x / Hh;
  int wave = threadIdx.x >> 5, lane = threadIdx.x & 31;
  int col = lane & 15, half = lane >> 4;

  __shared__ _Float16 pbuf[4][16 * 32];
  _Float16* P = pbuf[wave];

  const _Float16* Kbh = kb + (size_t)b * Ss * Dd + h * DHd;
  const _Float16* Vbh = vb + (size_t)b * Ss * Dd + h * DHd;
  const float*    mb  = mask + (size_t)b * Ss;

  int q0 = wave * 16;  // queries 0..63
  const _Float16* Qp = qb + ((size_t)b * Ss + q0) * Dd + h * DHd;
  v16h qa0 = load_afrag(Qp, Dd);
  v16h qa1 = load_afrag(Qp + 32, Dd);

  v8f o0 = {}, o1 = {}, o2 = {}, o3 = {};
  float mrun[8], lrun[8];
#pragma unroll
  for (int g = 0; g < 8; ++g) { mrun[g] = -1e30f; lrun[g] = 0.f; }

  for (int step = 0; step < Ss / 32; ++step) {
    int kp0 = step * 32 + col;
    int kp1 = kp0 + 16;
    v8f s0 = {}, s1 = {};
    {
      v16h bk;
      const _Float16* p = Kbh + (size_t)kp0 * Dd + 8 * half;
#pragma unroll
      for (int e = 0; e < 16; ++e) bk[e] = p[(e & 7) + ((e >> 3) << 4)];
      s0 = wmma32(qa0, bk, s0);
#pragma unroll
      for (int e = 0; e < 16; ++e) bk[e] = p[32 + (e & 7) + ((e >> 3) << 4)];
      s0 = wmma32(qa1, bk, s0);
      const _Float16* p1 = Kbh + (size_t)kp1 * Dd + 8 * half;
#pragma unroll
      for (int e = 0; e < 16; ++e) bk[e] = p1[(e & 7) + ((e >> 3) << 4)];
      s1 = wmma32(qa0, bk, s1);
#pragma unroll
      for (int e = 0; e < 16; ++e) bk[e] = p1[32 + (e & 7) + ((e >> 3) << 4)];
      s1 = wmma32(qa1, bk, s1);
    }
    float mv0 = mb[kp0], mv1 = mb[kp1];
#pragma unroll
    for (int g = 0; g < 8; ++g) {
      float sv0 = (mv0 > 0.f) ? s0[g] * 0.125f : NEGV;
      float sv1 = (mv1 > 0.f) ? s1[g] * 0.125f : NEGV;
      float tmax = fmaxf(sv0, sv1);
#pragma unroll
      for (int off = 8; off > 0; off >>= 1)
        tmax = fmaxf(tmax, __shfl_xor(tmax, off, 32));
      float mnew = fmaxf(mrun[g], tmax);
      float sc = __expf(mrun[g] - mnew);
      float p0 = __expf(sv0 - mnew), p1 = __expf(sv1 - mnew);
      float rs = p0 + p1;
#pragma unroll
      for (int off = 8; off > 0; off >>= 1) rs += __shfl_xor(rs, off, 32);
      lrun[g] = lrun[g] * sc + rs;
      mrun[g] = mnew;
      o0[g] *= sc; o1[g] *= sc; o2[g] *= sc; o3[g] *= sc;
      int prow = g + 8 * half;
      P[prow * 32 + col]      = (_Float16)p0;
      P[prow * 32 + 16 + col] = (_Float16)p1;
    }
    asm volatile("s_wait_dscnt 0" ::: "memory");
    v16h pa;
    {
      const _Float16* pp = P + (lane & 15) * 32 + 8 * half;
#pragma unroll
      for (int e = 0; e < 8; ++e) pa[e] = pp[e];
#pragma unroll
      for (int e = 0; e < 8; ++e) pa[8 + e] = pp[16 + e];
    }
    v16h v0, v1, v2, v3;
#pragma unroll
    for (int e = 0; e < 16; ++e) {
      int kp = step * 32 + 8 * half + (e & 7) + ((e >> 3) << 4);
      const _Float16* vp = Vbh + (size_t)kp * Dd + col;
      v0[e] = vp[0]; v1[e] = vp[16]; v2[e] = vp[32]; v3[e] = vp[48];
    }
    o0 = wmma32(pa, v0, o0);
    o1 = wmma32(pa, v1, o1);
    o2 = wmma32(pa, v2, o2);
    o3 = wmma32(pa, v3, o3);
  }

#pragma unroll
  for (int g = 0; g < 8; ++g) {
    float inv = (lrun[g] > 0.f) ? 1.f / lrun[g] : 0.f;
    int srow = q0 + g + 8 * half;
    _Float16* op = ctx16 + ((size_t)b * Ss + srow) * Dd + h * DHd;
    op[col]      = (_Float16)(o0[g] * inv);
    op[16 + col] = (_Float16)(o1[g] * inv);
    op[32 + col] = (_Float16)(o2[g] * inv);
    op[48 + col] = (_Float16)(o3[g] * inv);
  }
}

// ---------------------------------------------------------------------------
// Small helper kernels
// ---------------------------------------------------------------------------
// Convert K x N f32 weight -> N x K f16 ("transposed") for contiguous B-frags.
__global__ void cvt_transpose_f16(const float* __restrict__ src,
                                  _Float16* __restrict__ dst, int K, int N) {
  int idx = blockIdx.x * blockDim.x + threadIdx.x;
  if (idx < K * N) {
    int n = idx / K, k = idx % K;
    dst[idx] = (_Float16)src[(size_t)k * N + n];
  }
}

__global__ void embed_kernel(const int* __restrict__ ids,
                             const float* __restrict__ embed,
                             float* __restrict__ xf, _Float16* __restrict__ xb) {
  int row = blockIdx.x;
  int id = ids[row];
  const float* e = embed + (size_t)id * Dd;
  for (int j = threadIdx.x; j < Dd; j += blockDim.x) {
    float v = e[j];
    xf[(size_t)row * Dd + j] = v;
    xb[(size_t)row * Dd + j] = (_Float16)v;
  }
}

__global__ __launch_bounds__(256) void layernorm_kernel(
    const float* __restrict__ h, const float* __restrict__ gamma,
    const float* __restrict__ beta, float* __restrict__ hn,
    _Float16* __restrict__ hb) {
  int row = blockIdx.x;
  const float* hp = h + (size_t)row * Dd;
  float s = 0.f, ss = 0.f;
  for (int j = threadIdx.x; j < Dd; j += 256) {
    float v = hp[j];
    s += v; ss += v * v;
  }
  __shared__ float rs[256], rq[256];
  rs[threadIdx.x] = s; rq[threadIdx.x] = ss;
  __syncthreads();
  for (int o = 128; o > 0; o >>= 1) {
    if ((int)threadIdx.x < o) {
      rs[threadIdx.x] += rs[threadIdx.x + o];
      rq[threadIdx.x] += rq[threadIdx.x + o];
    }
    __syncthreads();
  }
  float mu = rs[0] / (float)Dd;
  float var = rq[0] / (float)Dd - mu * mu;
  float rstd = rsqrtf(var + 1e-5f);
  for (int j = threadIdx.x; j < Dd; j += 256) {
    float v = (hp[j] - mu) * rstd * gamma[j] + beta[j];
    hn[(size_t)row * Dd + j] = v;
    hb[(size_t)row * Dd + j] = (_Float16)v;
  }
}

__global__ __launch_bounds__(128) void heads_kernel(
    const float* __restrict__ hn, const float* __restrict__ Wa,
    const float* __restrict__ ba, const float* __restrict__ Wsp,
    const float* __restrict__ bs, float* __restrict__ ans,
    float* __restrict__ span) {
  int row = blockIdx.x;
  const float* hp = hn + (size_t)row * Dd;
  float acc[5] = {0, 0, 0, 0, 0};
  for (int j = threadIdx.x; j < Dd; j += 128) {
    float v = hp[j];
    acc[0] += v * Wa[j * 3 + 0];
    acc[1] += v * Wa[j * 3 + 1];
    acc[2] += v * Wa[j * 3 + 2];
    acc[3] += v * Wsp[j * 2 + 0];
    acc[4] += v * Wsp[j * 2 + 1];
  }
  __shared__ float red[5][128];
  for (int o = 0; o < 5; ++o) red[o][threadIdx.x] = acc[o];
  __syncthreads();
  for (int s = 64; s > 0; s >>= 1) {
    if ((int)threadIdx.x < s)
      for (int o = 0; o < 5; ++o)
        red[o][threadIdx.x] += red[o][threadIdx.x + s];
    __syncthreads();
  }
  if (threadIdx.x == 0) {
    ans[(size_t)row * 3 + 0]  = sigmoidf_(red[0][0] + ba[0]);
    ans[(size_t)row * 3 + 1]  = sigmoidf_(red[1][0] + ba[1]);
    ans[(size_t)row * 3 + 2]  = sigmoidf_(red[2][0] + ba[2]);
    span[(size_t)row * 2 + 0] = sigmoidf_(red[3][0] + bs[0]);
    span[(size_t)row * 2 + 1] = sigmoidf_(red[4][0] + bs[1]);
  }
}

__global__ void gather_kernel(const float* __restrict__ hn,
                              const int* __restrict__ idx,
                              _Float16* __restrict__ hr16) {
  int bm = blockIdx.x;               // [b*M + m]
  int b = bm / Mm;
  int s = idx[bm];
  const float* hp = hn + ((size_t)b * Ss + s) * Dd;
  for (int j = threadIdx.x; j < Dd; j += blockDim.x)
    hr16[(size_t)bm * Dd + j] = (_Float16)hp[j];
}

// ---------------------------------------------------------------------------
// Host launcher
// ---------------------------------------------------------------------------
extern "C" void kernel_launch(void* const* d_in, const int* in_sizes, int n_in,
                              void* d_out, int out_size, void* d_ws, size_t ws_size,
                              hipStream_t stream) {
  (void)in_sizes; (void)n_in; (void)out_size; (void)ws_size;
  const int*   ids   = (const int*)d_in[0];
  const float* amask = (const float*)d_in[1];
  const int*   hidx  = (const int*)d_in[2];
  const float* embed = (const float*)d_in[3];
  const float* Wq = (const float*)d_in[4];  const float* bq = (const float*)d_in[5];
  const float* Wk = (const float*)d_in[6];  const float* bk = (const float*)d_in[7];
  const float* Wv = (const float*)d_in[8];  const float* bv = (const float*)d_in[9];
  const float* Wo = (const float*)d_in[10]; const float* bo = (const float*)d_in[11];
  const float* gamma = (const float*)d_in[12]; const float* beta = (const float*)d_in[13];
  const float* Wp = (const float*)d_in[14]; const float* bp = (const float*)d_in[15];
  const float* Wa = (const float*)d_in[16]; const float* ba = (const float*)d_in[17];
  const float* Wsp = (const float*)d_in[18]; const float* bs = (const float*)d_in[19];

  float* out = (float*)d_out;
  const size_t NT   = (size_t)Bb * Ss;          // 8192 token rows
  const size_t ND   = NT * Dd;                  // 6291456
  const size_t OFF_ANS  = ND;
  const size_t OFF_SPAN = OFF_ANS + NT * 3;
  const size_t OFF_COND = OFF_SPAN + NT * 2;

  // -------- workspace carve (bytes), 256-aligned --------
  char* ws = (char*)d_ws;
  size_t off = 0;
  auto carve = [&](size_t bytes) {
    void* p = ws + off;
    off += (bytes + 255) & ~(size_t)255;
    return p;
  };
  float*    xf    = (float*)   carve(ND * 4);            // residual x (f32)
  float*    hbuf  = (float*)   carve(ND * 4);            // h = x + ctx@Wo + bo
  _Float16* xb    = (_Float16*)carve(ND * 2);
  _Float16* qb    = (_Float16*)carve(ND * 2);
  _Float16* kbuf  = (_Float16*)carve(ND * 2);
  _Float16* vbuf  = (_Float16*)carve(ND * 2);
  _Float16* ctx16 = (_Float16*)carve(ND * 2);
  _Float16* hb    = (_Float16*)carve(ND * 2);            // hn in f16
  _Float16* hr16  = (_Float16*)carve((size_t)Bb * Mm * Dd * 2);
  _Float16* tb    = (_Float16*)carve((size_t)Bb * Mm * Dd * 2);
  _Float16* w16q  = (_Float16*)carve((size_t)Dd * Dd * 2);   // all N x K
  _Float16* w16k  = (_Float16*)carve((size_t)Dd * Dd * 2);
  _Float16* w16v  = (_Float16*)carve((size_t)Dd * Dd * 2);
  _Float16* w16o  = (_Float16*)carve((size_t)Dd * Dd * 2);
  _Float16* w16p  = (_Float16*)carve((size_t)Dd * Dd * 2);

  const int nW = Dd * Dd;                        // 589824
  const int cvtGrid = (nW + 255) / 256;
  cvt_transpose_f16<<<cvtGrid, 256, 0, stream>>>(Wq, w16q, Dd, Dd);
  cvt_transpose_f16<<<cvtGrid, 256, 0, stream>>>(Wk, w16k, Dd, Dd);
  cvt_transpose_f16<<<cvtGrid, 256, 0, stream>>>(Wv, w16v, Dd, Dd);
  cvt_transpose_f16<<<cvtGrid, 256, 0, stream>>>(Wo, w16o, Dd, Dd);
  cvt_transpose_f16<<<cvtGrid, 256, 0, stream>>>(Wp, w16p, Dd, Dd);

  embed_kernel<<<(int)NT, 256, 0, stream>>>(ids, embed, xf, xb);

  // QKV projections (f16 WMMA, f32 accumulate); wave tile 32x64
  const int gemmGrid = ((int)(NT / 32) * (Dd / 64) + 7) / 8;   // 384
  gemm_f16_wmma<<<gemmGrid, 256, 0, stream>>>(xb, w16q, bq, nullptr,
      nullptr, qb,   (int)NT, Dd, Dd, 0);
  gemm_f16_wmma<<<gemmGrid, 256, 0, stream>>>(xb, w16k, bk, nullptr,
      nullptr, kbuf, (int)NT, Dd, Dd, 0);
  gemm_f16_wmma<<<gemmGrid, 256, 0, stream>>>(xb, w16v, bv, nullptr,
      nullptr, vbuf, (int)NT, Dd, Dd, 0);

  // attention
  attn_band_kernel<<<Bb * Hh * Cc, 256, 0, stream>>>(qb, kbuf, vbuf, amask, ctx16);
  attn_global_kernel<<<Bb * Hh, 128, 0, stream>>>(qb, kbuf, vbuf, amask, ctx16);

  // output projection + residual -> h
  gemm_f16_wmma<<<gemmGrid, 256, 0, stream>>>(ctx16, w16o, bo, xf,
      hbuf, nullptr, (int)NT, Dd, Dd, 0);

  // layernorm -> hn (f32, d_out) + f16 copy
  layernorm_kernel<<<(int)NT, 256, 0, stream>>>(hbuf, gamma, beta, out, hb);

  // ans / span heads (tiny, VALU)
  heads_kernel<<<(int)NT, 128, 0, stream>>>(out, Wa, ba, Wsp, bs,
                                            out + OFF_ANS, out + OFF_SPAN);

  // hr gather, t = relu(hr@Wp + bp)
  gather_kernel<<<Bb * Mm, 256, 0, stream>>>(out, hidx, hr16);
  const int gridWp = ((Bb * Mm / 32) * (Dd / 64) + 7) / 8;     // 24
  gemm_f16_wmma<<<gridWp, 256, 0, stream>>>(hr16, w16p, bp, nullptr,
      nullptr, tb, Bb * Mm, Dd, Dd, 1);

  // cond[b][c][q] = sigmoid(t[b,c,:] . hr[b,q,:]);
  // hr16 (M x K) is already the required N x K "transposed-B" layout.
  const int gridCond = ((Mm / 32) * (Mm / 64) + 7) / 8;        // 4
  for (int b = 0; b < Bb; ++b) {
    gemm_f16_wmma<<<gridCond, 256, 0, stream>>>(
        tb + (size_t)b * Mm * Dd, hr16 + (size_t)b * Mm * Dd,
        nullptr, nullptr,
        out + OFF_COND + (size_t)b * Mm * Mm, nullptr,
        Mm, Mm, Dd, 2);
  }
}
